// MultiScaleAttention5D_20658792694290
// MI455X (gfx1250) — compile-verified, matrix operand order checked by reading, not run on hardware
//
#include <hip/hip_runtime.h>

// ---------------------------------------------------------------------------
// Problem constants (from reference): B=2, C=256, T1=T2=L=2 (S=8), H=W=32
// ---------------------------------------------------------------------------
#define B_   2
#define C_   256
#define S_   8
#define N_   1024            // tokens per slice (H*W)
#define HN_  8               // heads
#define DH_  32              // head dim
#define SN_  (S_ * N_)       // 8192
#define CSN_ (C_ * SN_)      // 2097152 (per-batch elements)
#define TOK_ (B_ * SN_)      // 16384 tokens total
#define OC_  (3 * C_)        // 768 qkv output channels
#define QK_  (2 * C_)        // 512 (Q||K channels, token-major store)

typedef __attribute__((ext_vector_type(16))) _Float16 v16h;
typedef __attribute__((ext_vector_type(8)))  _Float16 v8h;
typedef __attribute__((ext_vector_type(8)))  float    v8f;

// ---------------------------------------------------------------------------
// Kernel 0: one-shot W f32 -> f16 conversion (W reused by all token blocks)
// ---------------------------------------------------------------------------
__global__ void wcvt_kernel(const float* __restrict__ w,
                            _Float16* __restrict__ wH) {
    const int idx = blockIdx.x * blockDim.x + threadIdx.x;  // 0..196607
    wH[idx] = (_Float16)w[idx];
}

// ---------------------------------------------------------------------------
// Kernel 1: QKV projection.  D[token, o] = sum_c x[c, token] * W[o, c] + b[o]
// Grid: 1024 token-blocks x 6 o-groups; 8 waves/block share one LDS x-tile.
//  - x tile [16 tok x 256 c] staged coalesced into LDS (f16, padded rows)
//  - A operand from LDS (2 x b128 ds reads per K-chunk)
//  - B operand = pre-converted f16 W rows, 2 x b128 global loads per K-chunk
// Q,K stored token-major: qkT[t*512 + o];  V stored d-major: vT[dv*16384 + t]
// ---------------------------------------------------------------------------
__global__ void qkv_gemm_kernel(const float* __restrict__ x,
                                const _Float16* __restrict__ wH,
                                const float* __restrict__ bias,
                                _Float16* __restrict__ qkT,
                                _Float16* __restrict__ vT) {
    __shared__ _Float16 xs[16][264];             // 264 = 256 + 8 halves pad

    const int tid  = threadIdx.x;
    const int lane = tid & 31;
    const int wid  = tid >> 5;
    const int ln = lane & 15, hi = lane >> 4;

    const int tokBlk = blockIdx.x / 6;           // 0..1023
    const int og     = blockIdx.x % 6;           // 0..5
    const int t0   = tokBlk * 16;
    const int bb   = t0 >> 13;                   // batch (token blocks never straddle)
    const int pos0 = t0 & (SN_ - 1);

    // ---- stage x tile: thread tid handles channel c = tid (16 coalesced floats)
    {
        const int c = tid;
        const float* xp = x + (size_t)bb * CSN_ + (size_t)c * SN_ + pos0;
        float4 f0 = *(const float4*)(xp + 0);
        float4 f1 = *(const float4*)(xp + 4);
        float4 f2 = *(const float4*)(xp + 8);
        float4 f3 = *(const float4*)(xp + 12);
        float vv[16] = { f0.x,f0.y,f0.z,f0.w, f1.x,f1.y,f1.z,f1.w,
                         f2.x,f2.y,f2.z,f2.w, f3.x,f3.y,f3.z,f3.w };
        #pragma unroll
        for (int k = 0; k < 16; ++k) xs[k][c] = (_Float16)vv[k];
    }
    __syncthreads();

    const int o0 = (og * 8 + wid) * 16;          // 0..752 step 16
    const _Float16* wlane = wH + (size_t)(o0 + ln) * C_;

    v8f acc = {};
    #pragma unroll
    for (int kc = 0; kc < 8; ++kc) {             // K = 256 in chunks of 32
        const int c0 = kc * 32;
        // A operand (X^T): lane row = token ln, contiguous c runs from LDS
        v8h alo = *(const v8h*)&xs[ln][c0 + hi * 8];
        v8h ahi = *(const v8h*)&xs[ln][c0 + 16 + hi * 8];
        // B operand (W^T): lane col = o, contiguous c runs from global (f16)
        v8h blo = *(const v8h*)(wlane + c0 + hi * 8);
        v8h bhi = *(const v8h*)(wlane + c0 + 16 + hi * 8);
        v16h a, bm;
        #pragma unroll
        for (int i = 0; i < 8; ++i) {
            a[i] = alo[i];  a[i + 8]  = ahi[i];
            bm[i] = blo[i]; bm[i + 8] = bhi[i];
        }
        acc = __builtin_amdgcn_wmma_f32_16x16x32_f16(
                false, a, false, bm, (short)0, acc, false, false);
    }

    const float bo = bias[o0 + ln];              // lane = output-channel column
    if (o0 < QK_) {
        // Q/K: token-major (coalesced b16 stores across lanes)
        #pragma unroll
        for (int j = 0; j < 8; ++j) {
            const int t = t0 + j + 8 * hi;       // D row = token
            qkT[(size_t)t * QK_ + o0 + ln] = (_Float16)(acc[j] + bo);
        }
    } else {
        // V: d-major; lane's 8 values are 8 consecutive tokens -> one b128 store
        const int dv = o0 + ln - QK_;            // 0..255
        v8h pk;
        #pragma unroll
        for (int j = 0; j < 8; ++j) pk[j] = (_Float16)(acc[j] + bo);
        *(v8h*)(vT + (size_t)dv * TOK_ + t0 + 8 * hi) = pk;
    }
}

// ---------------------------------------------------------------------------
// Kernel 2: flash attention. One wave handles 16 query tokens of one (b,h,s).
// Scores computed transposed: ST[key, query] so softmax columns are
// lane-local (+ lane^16 partner).  P tile feeds P*V WMMA lane-locally.
// All WMMA operands are contiguous 16B v8h loads (V via d-major vT).
// ---------------------------------------------------------------------------
__global__ void attn_kernel(const _Float16* __restrict__ qkT,
                            const _Float16* __restrict__ vT,
                            float* __restrict__ outw) {
    const int lane = threadIdx.x & 31;
    const int wid  = threadIdx.x >> 5;
    const int ln = lane & 15, hi = lane >> 4;

    const int wg    = blockIdx.x * 8 + wid;      // 0..8191
    const int slice = wg >> 6;                   // 0..127  (b,h,s)
    const int qblk  = wg & 63;                   // 64 query blocks of 16
    const int b   = slice >> 6;
    const int rem = slice & 63;
    const int h   = rem >> 3;
    const int s   = rem & 7;

    const int tok0 = b * SN_ + s * N_;
    const int oq = h * DH_;
    const int ok = C_ + h * DH_;
    const int q0 = qblk * 16;

    // Q operand (B-matrix): lane = query column; fold in 1/sqrt(d)
    v16h qv;
    {
        const _Float16* qp = qkT + (size_t)(tok0 + q0 + ln) * QK_ + oq;
        v8h lo  = *(const v8h*)(qp + hi * 8);
        v8h hi8 = *(const v8h*)(qp + 16 + hi * 8);
        const _Float16 sc = (_Float16)0.17677669529663687f;  // 1/sqrt(32)
        #pragma unroll
        for (int i = 0; i < 8; ++i) { qv[i] = lo[i] * sc; qv[i + 8] = hi8[i] * sc; }
    }

    // V row bases for this head (A-matrix rows = head-dim)
    const _Float16* v1row = vT + (size_t)(h * DH_ + ln) * TOK_ + tok0;
    const _Float16* v2row = v1row + (size_t)16 * TOK_;

    v8f O1 = {}, O2 = {};
    float run_max = -1e30f, run_sum = 0.0f;

    for (int m0 = 0; m0 < N_; m0 += 32) {        // 32 keys per iteration
        // K operand tiles (A-matrix rows = keys)
        const _Float16* kp0 = qkT + (size_t)(tok0 + m0 + ln) * QK_ + ok;
        const _Float16* kp1 = kp0 + 16 * QK_;
        // prefetch next iteration's K/V rows (global_prefetch_b8)
        __builtin_prefetch(kp0 + 32 * QK_, 0, 1);
        __builtin_prefetch(v1row + m0 + 32, 0, 1);

        v16h ak0, ak1, av1, av2;
        {
            v8h l0 = *(const v8h*)(kp0 + hi * 8);
            v8h u0 = *(const v8h*)(kp0 + 16 + hi * 8);
            v8h l1 = *(const v8h*)(kp1 + hi * 8);
            v8h u1 = *(const v8h*)(kp1 + 16 + hi * 8);
            // V tiles: K dim = tokens, contiguous runs in vT
            v8h a1l = *(const v8h*)(v1row + m0 + hi * 8);
            v8h a1u = *(const v8h*)(v1row + m0 + 16 + hi * 8);
            v8h a2l = *(const v8h*)(v2row + m0 + hi * 8);
            v8h a2u = *(const v8h*)(v2row + m0 + 16 + hi * 8);
            #pragma unroll
            for (int i = 0; i < 8; ++i) {
                ak0[i] = l0[i];  ak0[i + 8] = u0[i];
                ak1[i] = l1[i];  ak1[i + 8] = u1[i];
                av1[i] = a1l[i]; av1[i + 8] = a1u[i];
                av2[i] = a2l[i]; av2[i + 8] = a2u[i];
            }
        }
        v8f z = {};
        v8f st0 = __builtin_amdgcn_wmma_f32_16x16x32_f16(
                      false, ak0, false, qv, (short)0, z, false, false);
        v8f st1 = __builtin_amdgcn_wmma_f32_16x16x32_f16(
                      false, ak1, false, qv, (short)0, z, false, false);

        // online softmax per query column
        float pm = st0[0];
        #pragma unroll
        for (int j = 0; j < 8; ++j) { pm = fmaxf(pm, st0[j]); pm = fmaxf(pm, st1[j]); }
        pm = fmaxf(pm, __shfl_xor(pm, 16));
        const float nm   = fmaxf(run_max, pm);
        const float corr = __expf(run_max - nm);

        float p0[8], p1[8], ps = 0.0f;
        #pragma unroll
        for (int j = 0; j < 8; ++j) {
            p0[j] = __expf(st0[j] - nm);
            p1[j] = __expf(st1[j] - nm);
            ps += p0[j] + p1[j];
        }
        ps += __shfl_xor(ps, 16);
        run_sum = run_sum * corr + ps;
        run_max = nm;
        #pragma unroll
        for (int j = 0; j < 8; ++j) { O1[j] *= corr; O2[j] *= corr; }

        // P as B-operand: lane-local repack (D-layout rows -> B halves)
        v16h pB;
        #pragma unroll
        for (int j = 0; j < 8; ++j) {
            pB[j]     = (_Float16)p0[j];
            pB[j + 8] = (_Float16)p1[j];
        }

        O1 = __builtin_amdgcn_wmma_f32_16x16x32_f16(
                 false, av1, false, pB, (short)0, O1, false, false);
        O2 = __builtin_amdgcn_wmma_f32_16x16x32_f16(
                 false, av2, false, pB, (short)0, O2, false, false);
    }

    const float inv = 1.0f / run_sum;
    const size_t base = (size_t)b * CSN_ + s * N_ + q0 + ln;
    #pragma unroll
    for (int j = 0; j < 8; ++j) {
        const int dd = j + 8 * hi;
        outw[base + (size_t)(h * DH_ + dd) * SN_]      = O1[j] * inv;
        outw[base + (size_t)(h * DH_ + 16 + dd) * SN_] = O2[j] * inv;
    }
}

// ---------------------------------------------------------------------------
// Kernel 3: per (b, group) mean / rstd of y = attn_out + x
// ---------------------------------------------------------------------------
__global__ void gstats_kernel(const float* __restrict__ x,
                              const float* __restrict__ outw,
                              float* __restrict__ stats) {
    const int bg = blockIdx.x;               // 0..15
    const int b = bg >> 3, g = bg & 7;
    const int tid = threadIdx.x;
    const int GE = (C_ / 8) * SN_;           // 262144 elems per group

    float s = 0.0f, ss = 0.0f;
    for (int e = tid; e < GE; e += blockDim.x) {
        const int c   = g * (C_ / 8) + (e >> 13);
        const int pos = e & (SN_ - 1);
        const size_t idx = (size_t)b * CSN_ + (size_t)c * SN_ + pos;
        const float y = outw[idx] + x[idx];
        s += y; ss += y * y;
    }
    __shared__ float sh1[256], sh2[256];
    sh1[tid] = s; sh2[tid] = ss;
    __syncthreads();
    for (int off = 128; off > 0; off >>= 1) {
        if (tid < off) { sh1[tid] += sh1[tid + off]; sh2[tid] += sh2[tid + off]; }
        __syncthreads();
    }
    if (tid == 0) {
        const float mean = sh1[0] / (float)GE;
        const float var  = sh2[0] / (float)GE - mean * mean;
        stats[2 * bg]     = mean;
        stats[2 * bg + 1] = rsqrtf(var + 1e-5f);
    }
}

// ---------------------------------------------------------------------------
// Kernel 4: normalize + affine
// ---------------------------------------------------------------------------
__global__ void finalize_kernel(const float* __restrict__ x,
                                const float* __restrict__ outw,
                                const float* __restrict__ stats,
                                const float* __restrict__ gamma,
                                const float* __restrict__ beta,
                                float* __restrict__ out) {
    const size_t idx = (size_t)blockIdx.x * blockDim.x + threadIdx.x;
    if (idx >= (size_t)B_ * CSN_) return;
    const int b = (int)(idx / CSN_);
    const int c = (int)((idx / SN_) % C_);
    const int g = c / (C_ / 8);
    const float mean = stats[2 * (b * 8 + g)];
    const float rstd = stats[2 * (b * 8 + g) + 1];
    const float y = outw[idx] + x[idx];
    out[idx] = (y - mean) * rstd * gamma[c] + beta[c];
}

// ---------------------------------------------------------------------------
extern "C" void kernel_launch(void* const* d_in, const int* in_sizes, int n_in,
                              void* d_out, int out_size, void* d_ws, size_t ws_size,
                              hipStream_t stream) {
    const float* x     = (const float*)d_in[0];
    const float* qkv_w = (const float*)d_in[1];
    const float* qkv_b = (const float*)d_in[2];
    const float* gamma = (const float*)d_in[3];
    const float* beta  = (const float*)d_in[4];
    float* out = (float*)d_out;

    char* ws = (char*)d_ws;
    _Float16* qkT   = (_Float16*)ws;                           // 16384*512*2 = 16777216 B
    _Float16* vT    = (_Float16*)(ws + 16777216);              // 256*16384*2 =  8388608 B
    float*    outw  = (float*)(ws + 16777216 + 8388608);       // 4194304*4   = 16777216 B
    float*    stats = (float*)(ws + 41943040);                 // 128 B
    _Float16* wH    = (_Float16*)(ws + 41943040 + 256);        // 768*256*2   =   393216 B

    // 0) One-shot W -> f16 (reused by every token block)
    wcvt_kernel<<<(OC_ * C_) / 256, 256, 0, stream>>>(qkv_w, wH);
    // 1) QKV GEMM: 1024 token-blocks x 6 o-groups, 8 wave-tiles per block
    qkv_gemm_kernel<<<6144, 256, 0, stream>>>(x, wH, qkv_b, qkT, vT);
    // 2) Attention: 128 slices * 64 query blocks = 8192 wave jobs
    attn_kernel<<<1024, 256, 0, stream>>>(qkT, vT, outw);
    // 3) GroupNorm statistics: one block per (b, group)
    gstats_kernel<<<16, 256, 0, stream>>>(x, outw, stats);
    // 4) Normalize + affine over all 4,194,304 elements
    finalize_kernel<<<(B_ * CSN_ + 255) / 256, 256, 0, stream>>>(
        x, outw, stats, gamma, beta, out);
}
